// DecoderBlock_12910671691782
// MI455X (gfx1250) — matrix operand order repaired and checked by reading, not compiled, not run on hardware
//
#include <hip/hip_runtime.h>
#include <hip/hip_bf16.h>

// MI455X (gfx1250, wave32) decoder block.
// All matmuls via v_wmma_f32_16x16x32_bf16 (f32 accumulate).
// Fragment loads are explicit 128-bit vector loads matching the documented
// CDNA5 WMMA VGPR layouts; inner loops are software-pipelined so loads for
// step k+1 are in flight while WMMAs for step k execute.

typedef __bf16 bf16;
typedef __attribute__((ext_vector_type(16))) __bf16 v16bf;
typedef __attribute__((ext_vector_type(8)))  float  v8f;
typedef __attribute__((ext_vector_type(4)))  unsigned int v4u;

#define NEG_INF_F (-__builtin_inff())

union frag_u { v16bf bf; v4u u[2]; };

__device__ __forceinline__ v8f wmma_bf16(v16bf a, v16bf b, v8f c) {
  return __builtin_amdgcn_wmma_f32_16x16x32_bf16(
      /*neg_a=*/false, a, /*neg_b=*/false, b,
      /*c_mod=*/(short)0, c, /*reuse_a=*/false, /*reuse_b=*/false);
}

// A-matrix 16x32 bf16 fragment (ISA 7.12.2): lane row m = lane&15,
// half=lane>>4. Elements are two contiguous 16B runs:
//   f[0..7]  = arow[k0 + 8*half      .. +7]
//   f[8..15] = arow[k0 + 16 + 8*half .. +7]
__device__ __forceinline__ v16bf a_frag(const bf16* arow, int k0, int half) {
  frag_u f;
  f.u[0] = *(const v4u*)(arow + k0 + 8 * half);
  f.u[1] = *(const v4u*)(arow + k0 + 16 + 8 * half);
  return f.bf;
}

// B-matrix 32x16 bf16 fragment: lane col n = lane&15, same K order as A:
//   f[0..7] = bcol[k0 + 8*half .. +7], f[8..15] = bcol[k0 + 16 + 8*half .. +7].
__device__ __forceinline__ v16bf b_frag(const bf16* bcol, int k0, int half) {
  frag_u f;
  const bf16* p = bcol + k0 + 8 * half;
  f.u[0] = *(const v4u*)(p);
  f.u[1] = *(const v4u*)(p + 16);
  return f.bf;
}

// ---------------- LayerNorm: one row (C=1024) per 256-thread block -> bf16 ----
__global__ void ln_kernel(const float* __restrict__ x, const float* __restrict__ g,
                          const float* __restrict__ bb, bf16* __restrict__ y) {
  const int Cc = 1024;
  int row = blockIdx.x, tid = threadIdx.x;
  const float* xr = x + (size_t)row * Cc;
  float4 v = ((const float4*)xr)[tid];
  __shared__ float r1[256], r2[256];
  r1[tid] = v.x + v.y + v.z + v.w;
  r2[tid] = v.x * v.x + v.y * v.y + v.z * v.z + v.w * v.w;
  __syncthreads();
  for (int off = 128; off > 0; off >>= 1) {
    if (tid < off) { r1[tid] += r1[tid + off]; r2[tid] += r2[tid + off]; }
    __syncthreads();
  }
  float mu = r1[0] * (1.0f / Cc);
  float var = r2[0] * (1.0f / Cc) - mu * mu;
  float rstd = rsqrtf(var + 1e-5f);
  float4 gg = ((const float4*)g)[tid];
  float4 bv = ((const float4*)bb)[tid];
  union { bf16 h[4]; uint2 u; } o;
  o.h[0] = (bf16)((v.x - mu) * rstd * gg.x + bv.x);
  o.h[1] = (bf16)((v.y - mu) * rstd * gg.y + bv.y);
  o.h[2] = (bf16)((v.z - mu) * rstd * gg.z + bv.z);
  o.h[3] = (bf16)((v.w - mu) * rstd * gg.w + bv.w);
  ((uint2*)(y + (size_t)row * Cc))[tid] = o.u;
}

// ---------------- Weight transpose+convert: W[K,N] f32 -> Wt[N,K] bf16 --------
__global__ void wtrans_kernel(const float* __restrict__ W, bf16* __restrict__ Wt,
                              int K, int N) {
  __shared__ bf16 buf[32][33];
  int n0 = blockIdx.x * 32, k0 = blockIdx.y * 32;
  for (int idx = threadIdx.x; idx < 1024; idx += 256) {
    int n = idx & 31, k = idx >> 5;
    buf[k][n] = (bf16)W[(size_t)(k0 + k) * N + n0 + n];
  }
  __syncthreads();
  for (int idx = threadIdx.x; idx < 1024; idx += 256) {
    int k = idx & 31, n = idx >> 5;
    Wt[(size_t)(n0 + n) * K + k0 + k] = buf[k][n];
  }
}

// ---------------- V transpose: V[(b*S+s)*vs + h*64 + d] -> Vt[(b*H+h)*64+d][S] -
__global__ void vtrans_kernel(const bf16* __restrict__ V, int vs,
                              bf16* __restrict__ Vt, int S) {
  __shared__ bf16 buf[32][65];
  int b = blockIdx.z, h = blockIdx.y, s0 = blockIdx.x * 32;
  const bf16* vin = V + ((size_t)b * S + s0) * vs + h * 64;
  for (int idx = threadIdx.x; idx < 32 * 64; idx += 256) {
    int d = idx & 63, s = idx >> 6;
    buf[s][d] = vin[(size_t)s * vs + d];
  }
  __syncthreads();
  bf16* vout = Vt + (size_t)(b * 16 + h) * 64 * S + s0;
  for (int idx = threadIdx.x; idx < 32 * 64; idx += 256) {
    int s = idx & 31, d = idx >> 5;
    vout[(size_t)d * S + s] = buf[s][d];
  }
}

// ---------------- GEMM: out[M,N] = A[M,K](bf16) * Wt[N,K]^T (+bias,+res,relu) --
// Block = 256 threads = 8 waves. Wave tile = 16(M) x 64(N). Double-buffered:
// fragments for k+32 are loaded while WMMAs for k execute.
template <bool BIAS, bool RELU, bool RES, bool OBF16>
__global__ void __launch_bounds__(256, 1)
gemm_kernel(const bf16* __restrict__ A, const bf16* __restrict__ Wt,
            const float* __restrict__ bias,
            const float* __restrict__ res, void* __restrict__ outp,
            int M, int N, int K) {
  int lane = threadIdx.x & 31;
  int wave = threadIdx.x >> 5;
  int ln = lane & 15, half = lane >> 4;
  int wm = wave & 1, wn = wave >> 1;
  int m0 = blockIdx.y * 32 + wm * 16;
  int n0 = blockIdx.x * 256 + wn * 64;

  const bf16* arow = A + (size_t)(m0 + ln) * K;
  const bf16* bcol[4];
#pragma unroll
  for (int j = 0; j < 4; ++j) bcol[j] = Wt + (size_t)(n0 + j * 16 + ln) * K;

  v8f z = {0.f, 0.f, 0.f, 0.f, 0.f, 0.f, 0.f, 0.f};
  v8f acc[4] = {z, z, z, z};

  // prologue: fragments for k = 0
  v16bf af = a_frag(arow, 0, half);
  v16bf bf[4];
#pragma unroll
  for (int j = 0; j < 4; ++j) bf[j] = b_frag(bcol[j], 0, half);

  // steady state: load k0+32 while multiplying k0
  for (int k0 = 0; k0 + 32 < K; k0 += 32) {
    __builtin_prefetch(arow + k0 + 256, 0, 3);  // global_prefetch_b8 (speculative)
    v16bf afn = a_frag(arow, k0 + 32, half);
    v16bf bfn[4];
#pragma unroll
    for (int j = 0; j < 4; ++j) bfn[j] = b_frag(bcol[j], k0 + 32, half);
#pragma unroll
    for (int j = 0; j < 4; ++j) acc[j] = wmma_bf16(af, bf[j], acc[j]);
    af = afn;
#pragma unroll
    for (int j = 0; j < 4; ++j) bf[j] = bfn[j];
  }
  // epilogue step (last 32-deep slice)
#pragma unroll
  for (int j = 0; j < 4; ++j) acc[j] = wmma_bf16(af, bf[j], acc[j]);

#pragma unroll
  for (int j = 0; j < 4; ++j) {
    int n = n0 + j * 16 + ln;
    float bv = BIAS ? bias[n] : 0.f;
#pragma unroll
    for (int r = 0; r < 8; ++r) {
      int m = m0 + r + 8 * half;  // C/D layout: lane n=l&15, row = r + 8*(l>>4)
      float v = acc[j][r] + bv;
      if (RES) v += res[(size_t)m * N + n];
      if (RELU) v = fmaxf(v, 0.f);
      if (OBF16) ((bf16*)outp)[(size_t)m * N + n] = (bf16)v;
      else       ((float*)outp)[(size_t)m * N + n] = v;
    }
  }
}

// ---------------- Flash attention: 1 wave / 16 query rows; 32 kv cols / step --
// All fragments of a step are loaded (batched b128 clauses) before the WMMAs
// that consume them. P re-shaped to A-frag layout through 1KB of LDS.
__global__ void __launch_bounds__(32, 1)
attn_kernel(const bf16* __restrict__ qb, const bf16* __restrict__ kb,
            const bf16* __restrict__ vt, int qs, int ks,
            int Tq, int Skv, int causal,
            const unsigned char* __restrict__ maskb, float scale,
            bf16* __restrict__ outb, int os) {
  int b = blockIdx.z, h = blockIdx.y;
  int t0 = blockIdx.x * 16;
  int lane = threadIdx.x;
  int ln = lane & 15, half = lane >> 4;

  const bf16* qrow = qb + ((size_t)b * Tq + t0 + ln) * qs + h * 64;
  v16bf qf0 = a_frag(qrow, 0, half);
  v16bf qf1 = a_frag(qrow, 32, half);
  const unsigned char* mrow = maskb ? (maskb + (size_t)b * Skv) : nullptr;
  const bf16* kbase = kb + (size_t)b * Skv * ks + h * 64;
  const bf16* vrow[4];
#pragma unroll
  for (int j = 0; j < 4; ++j)
    vrow[j] = vt + ((size_t)(b * 16 + h) * 64 + j * 16 + ln) * Skv;

  v8f z = {0.f, 0.f, 0.f, 0.f, 0.f, 0.f, 0.f, 0.f};
  v8f oacc[4] = {z, z, z, z};
  float mrun[8], lrun[8];
#pragma unroll
  for (int r = 0; r < 8; ++r) { mrun[r] = NEG_INF_F; lrun[r] = 0.f; }

  __shared__ alignas(16) bf16 pbuf[16 * 32];

  int send = causal ? (t0 + 16) : Skv;
  for (int s0 = 0; s0 < send; s0 += 32) {
    // ---- QK^T: batch all 4 K-fragments, then 4 WMMAs ----
    const bf16* krow0 = kbase + (size_t)(s0 + ln) * ks;
    const bf16* krow1 = kbase + (size_t)(s0 + 16 + ln) * ks;
    v16bf kf00 = b_frag(krow0, 0, half);
    v16bf kf01 = b_frag(krow0, 32, half);
    v16bf kf10 = b_frag(krow1, 0, half);
    v16bf kf11 = b_frag(krow1, 32, half);
    __builtin_prefetch(kbase + (size_t)(s0 + 32 + ln) * ks, 0, 3);  // speculative
    v8f st[2];
    {
      v8f a0 = z, a1 = z;
      a0 = wmma_bf16(qf0, kf00, a0);
      a1 = wmma_bf16(qf0, kf10, a1);
      a0 = wmma_bf16(qf1, kf01, a0);
      a1 = wmma_bf16(qf1, kf11, a1);
      st[0] = a0; st[1] = a1;
    }
    // ---- scale + causal / key-padding mask (col = ln, row = r+8*half) ----
#pragma unroll
    for (int sub = 0; sub < 2; ++sub) {
      int sg = s0 + sub * 16 + ln;
      bool pad = mrow && mrow[sg];
#pragma unroll
      for (int r = 0; r < 8; ++r) {
        float x = st[sub][r] * scale;
        int tr = t0 + r + 8 * half;
        if ((causal && sg > tr) || pad) x = NEG_INF_F;
        st[sub][r] = x;
      }
    }
    // ---- online softmax per row (row spread over 16 lanes of same half) ----
#pragma unroll
    for (int r = 0; r < 8; ++r) {
      float mx = fmaxf(st[0][r], st[1][r]);
#pragma unroll
      for (int off = 1; off < 16; off <<= 1) mx = fmaxf(mx, __shfl_xor(mx, off, 32));
      float mnew = fmaxf(mrun[r], mx);
      float corr = __expf(mrun[r] - mnew);
      float p0 = __expf(st[0][r] - mnew);
      float p1 = __expf(st[1][r] - mnew);
      float rs = p0 + p1;
#pragma unroll
      for (int off = 1; off < 16; off <<= 1) rs += __shfl_xor(rs, off, 32);
      lrun[r] = lrun[r] * corr + rs;
      mrun[r] = mnew;
#pragma unroll
      for (int j = 0; j < 4; ++j) oacc[j][r] *= corr;
      pbuf[(r + 8 * half) * 32 + ln]      = (bf16)p0;
      pbuf[(r + 8 * half) * 32 + 16 + ln] = (bf16)p1;
    }
    __syncthreads();
    // ---- P @ V: batch P + 4 V-fragments, then 4 WMMAs ----
    v16bf pf = a_frag(&pbuf[ln * 32], 0, half);  // 2x ds_load_b128
    v16bf vf[4];
#pragma unroll
    for (int j = 0; j < 4; ++j) vf[j] = b_frag(vrow[j], s0, half);
#pragma unroll
    for (int j = 0; j < 4; ++j) oacc[j] = wmma_bf16(pf, vf[j], oacc[j]);
    __syncthreads();
  }

#pragma unroll
  for (int j = 0; j < 4; ++j) {
#pragma unroll
    for (int r = 0; r < 8; ++r) {
      int m = t0 + r + 8 * half;
      float v = oacc[j][r] / lrun[r];
      outb[((size_t)b * Tq + m) * os + h * 64 + j * 16 + ln] = (bf16)v;
    }
  }
}

// -----------------------------------------------------------------------------
extern "C" void kernel_launch(void* const* d_in, const int* in_sizes, int n_in,
                              void* d_out, int out_size, void* d_ws, size_t ws_size,
                              hipStream_t stream) {
  (void)in_sizes; (void)n_in; (void)out_size; (void)ws_size;
  constexpr int Bv = 16, T = 448, J = 448, C = 1024, H = 16;
  constexpr int M = Bv * T;  // 7168 rows (== Bv*J)
  const float scale = 1.0f / 32.0f;  // C^-0.5

  const float* tgt      = (const float*)d_in[0];
  const float* src      = (const float*)d_in[1];
  const float* sa_wqkv  = (const float*)d_in[2];
  const float* sa_wproj = (const float*)d_in[3];
  const float* sa_bproj = (const float*)d_in[4];
  const float* ed_wkv   = (const float*)d_in[5];
  const float* ed_wq    = (const float*)d_in[6];
  const float* ed_wproj = (const float*)d_in[7];
  const float* ed_bproj = (const float*)d_in[8];
  const float* ff_w1    = (const float*)d_in[9];
  const float* ff_b1    = (const float*)d_in[10];
  const float* ff_w2    = (const float*)d_in[11];
  const float* ff_b2    = (const float*)d_in[12];
  const float* ln1_g = (const float*)d_in[13];
  const float* ln1_b = (const float*)d_in[14];
  const float* ln2_g = (const float*)d_in[15];
  const float* ln2_b = (const float*)d_in[16];
  const float* ln3_g = (const float*)d_in[17];
  const float* ln3_b = (const float*)d_in[18];
  const unsigned char* maskp = (const unsigned char*)d_in[19];
  float* out = (float*)d_out;

  char* w = (char*)d_ws;
  auto alloc = [&](size_t bytes) {
    void* p = (void*)w;
    w += (bytes + 255) & ~(size_t)255;
    return p;
  };
  bf16* wqkvT   = (bf16*)alloc((size_t)3 * C * C * 2);
  bf16* wprojT  = (bf16*)alloc((size_t)C * C * 2);
  bf16* wkvT    = (bf16*)alloc((size_t)2 * C * C * 2);
  bf16* wqT     = (bf16*)alloc((size_t)C * C * 2);
  bf16* wedprjT = (bf16*)alloc((size_t)C * C * 2);
  bf16* w1T     = (bf16*)alloc((size_t)4 * C * C * 2);
  bf16* w2T     = (bf16*)alloc((size_t)4 * C * C * 2);
  bf16* lnA  = (bf16*)alloc((size_t)M * C * 2);       // ln1(tgt) then ln3(tgt)
  bf16* lnB  = (bf16*)alloc((size_t)M * C * 2);       // ln2(result)
  bf16* lnC  = (bf16*)alloc((size_t)M * C * 2);       // ln2(src)
  bf16* qkv  = (bf16*)alloc((size_t)M * 3 * C * 2);   // fused q|k|v
  bf16* kv   = (bf16*)alloc((size_t)M * 2 * C * 2);   // fused k2|v2
  bf16* q2   = (bf16*)alloc((size_t)M * C * 2);
  bf16* attn = (bf16*)alloc((size_t)M * C * 2);       // merged attn out (reused)
  bf16* vtb  = (bf16*)alloc((size_t)M * C * 2);       // transposed V (reused)
  bf16* hbuf = (bf16*)alloc((size_t)M * 4 * C * 2);   // FFN hidden
  float* res1 = (float*)alloc((size_t)M * C * 4);     // tgt + sa_out

  auto wtr = [&](const float* W, bf16* Wt, int K, int N) {
    wtrans_kernel<<<dim3(N / 32, K / 32), 256, 0, stream>>>(W, Wt, K, N);
  };
  // --- weight convert/transpose (bf16, [N,K]) ---
  wtr(sa_wqkv, wqkvT, C, 3 * C);
  wtr(sa_wproj, wprojT, C, C);
  wtr(ed_wkv, wkvT, C, 2 * C);
  wtr(ed_wq, wqT, C, C);
  wtr(ed_wproj, wedprjT, C, C);
  wtr(ff_w1, w1T, C, 4 * C);
  wtr(ff_w2, w2T, 4 * C, C);

  dim3 blk(256);
  // --- self attention ---
  ln_kernel<<<M, blk, 0, stream>>>(tgt, ln1_g, ln1_b, lnA);
  gemm_kernel<false, false, false, true>
      <<<dim3(3 * C / 256, M / 32), blk, 0, stream>>>(lnA, wqkvT, nullptr, nullptr,
                                                      qkv, M, 3 * C, C);
  vtrans_kernel<<<dim3(T / 32, H, Bv), blk, 0, stream>>>(qkv + 2 * C, 3 * C, vtb, T);
  attn_kernel<<<dim3(T / 16, H, Bv), 32, 0, stream>>>(
      qkv, qkv + C, vtb, 3 * C, 3 * C, T, T, 1, maskp, scale, attn, C);
  gemm_kernel<true, false, true, false>
      <<<dim3(C / 256, M / 32), blk, 0, stream>>>(attn, wprojT, sa_bproj, tgt,
                                                  res1, M, C, C);
  // --- cross attention (ln2 on both result and src) ---
  ln_kernel<<<M, blk, 0, stream>>>(res1, ln2_g, ln2_b, lnB);
  ln_kernel<<<M, blk, 0, stream>>>(src, ln2_g, ln2_b, lnC);
  gemm_kernel<false, false, false, true>
      <<<dim3(2 * C / 256, M / 32), blk, 0, stream>>>(lnC, wkvT, nullptr, nullptr,
                                                      kv, M, 2 * C, C);
  gemm_kernel<false, false, false, true>
      <<<dim3(C / 256, M / 32), blk, 0, stream>>>(lnB, wqT, nullptr, nullptr,
                                                  q2, M, C, C);
  vtrans_kernel<<<dim3(J / 32, H, Bv), blk, 0, stream>>>(kv + C, 2 * C, vtb, J);
  attn_kernel<<<dim3(T / 16, H, Bv), 32, 0, stream>>>(
      q2, kv, vtb, C, 2 * C, T, J, 0, nullptr, scale, attn, C);
  gemm_kernel<true, false, true, false>
      <<<dim3(C / 256, M / 32), blk, 0, stream>>>(attn, wedprjT, ed_bproj, res1,
                                                  out, M, C, C);
  // --- FFN on ln3(tgt), residual into d_out ---
  ln_kernel<<<M, blk, 0, stream>>>(tgt, ln3_g, ln3_b, lnA);
  gemm_kernel<true, true, false, true>
      <<<dim3(4 * C / 256, M / 32), blk, 0, stream>>>(lnA, w1T, ff_b1, nullptr,
                                                      hbuf, M, 4 * C, C);
  gemm_kernel<true, false, true, false>
      <<<dim3(C / 256, M / 32), blk, 0, stream>>>(hbuf, w2T, ff_b2, out,
                                                  out, M, C, 4 * C);
}
